// TransformerBlock_77051713290546
// MI455X (gfx1250) — compile-verified, matrix operand order checked by reading, not run on hardware
//
#include <hip/hip_runtime.h>
#include <hip/hip_bf16.h>
#include <cstdint>
#include <cstddef>

// ---------------------------------------------------------------- types
typedef __bf16 bf16_t;
typedef __attribute__((ext_vector_type(16))) __bf16 bf16x16;
typedef __attribute__((ext_vector_type(8)))  float  f32x8;

__device__ inline f32x8 zero8() {
    f32x8 z;
#pragma unroll
    for (int i = 0; i < 8; ++i) z[i] = 0.f;
    return z;
}

__device__ inline f32x8 wmma_bf16(bf16x16 a, bf16x16 b, f32x8 c) {
    // D = A(16x32) * B(32x16) + C, f32 accumulate
    return __builtin_amdgcn_wmma_f32_16x16x32_bf16(
        /*neg_a=*/false, a, /*neg_b=*/false, b,
        /*c_mod=*/(short)0, c, /*reuse_a=*/false, /*reuse_b=*/false);
}

// CDNA5 async global->LDS copy (no VGPR staging), tracked by ASYNCcnt.
// lds_off = wave-relative LDS byte offset (low 32 bits of generic pointer).
__device__ inline void async_ld16(uint32_t lds_off, uint64_t gaddr) {
    asm volatile("global_load_async_to_lds_b128 %0, %1, off"
                 :: "v"(lds_off), "v"(gaddr) : "memory");
}
__device__ inline void wait_async_le4() { asm volatile("s_wait_asynccnt 0x4" ::: "memory"); }
__device__ inline void wait_async_le0() { asm volatile("s_wait_asynccnt 0x0" ::: "memory"); }

// Fragment loader for the CDNA5 16-bit A/B layout.
// Matrix stored in LDS as rows x stride(bf16), row = M (or N for B^T).
// lane<16: elements 0..7 = K 0..7, 8..15 = K 16..23  (byte offsets 0, 32)
// lane>=16: elements 0..7 = K 8..15, 8..15 = K 24..31 (byte offsets 16, 48)
__device__ inline bf16x16 ld_frag(const bf16_t* base, int row0, int stride, int lane) {
    const int m = lane & 15;
    const int khalf = (lane >> 4) * 8;            // 0 or 8
    const bf16_t* p = base + (size_t)(row0 + m) * stride + khalf;
    union { bf16x16 v; uint4 q[2]; } u;
    u.q[0] = *(const uint4*)(p);                  // K khalf .. khalf+7
    u.q[1] = *(const uint4*)(p + 16);             // K khalf+16 .. khalf+23
    return u.v;
}

// ---------------------------------------------------------------- GEMM
// C(MxN) = A(MxK, bf16, row-major) * Bt(NxK, bf16, row-major)^T  [+ epilogue]
// Double-buffered LDS, async global->LDS pipeline.
#define BM 128
#define BN 128
#define BK 32
#define LDSK 40   // BK + 8 pad (multiple of 8 keeps 16B alignment)

enum { EPI_F32 = 0, EPI_F32_BIAS = 1, EPI_BF16 = 2, EPI_RES_ADD = 3, EPI_ACCUM = 4 };

template <int EPI>
__global__ __launch_bounds__(256) void gemm_bf16_kernel(
    const bf16_t* __restrict__ A, const bf16_t* __restrict__ Bt,
    int M, int N, int K,
    float* __restrict__ outF, bf16_t* __restrict__ outB,
    const float* __restrict__ bias, const float* __restrict__ res)
{
    __shared__ __align__(16) bf16_t As[2][BM * LDSK];
    __shared__ __align__(16) bf16_t Bs[2][BN * LDSK];

    const int tid  = threadIdx.x;
    const int lane = tid & 31;
    const int w    = tid >> 5;               // 8 waves
    const int bm   = blockIdx.y * BM;
    const int bn   = blockIdx.x * BN;
    const int wm   = (w >> 2) * 64;          // wave tile: 64 x 32
    const int wn   = (w & 3) * 32;

    f32x8 acc[4][2];
#pragma unroll
    for (int i = 0; i < 4; ++i)
#pragma unroll
        for (int j = 0; j < 2; ++j) acc[i][j] = zero8();

    // cooperative async copies: 2 threads per row, 2x16B each, per matrix
    const int lr = tid >> 1;                 // 0..127
    const int lc = (tid & 1) * 16;           // 0 or 16
    const uint64_t gA = (uint64_t)(uintptr_t)(A  + (size_t)(bm + lr) * K + lc);
    const uint64_t gB = (uint64_t)(uintptr_t)(Bt + (size_t)(bn + lr) * K + lc);
    const uint32_t lA[2] = { (uint32_t)(uintptr_t)&As[0][lr * LDSK + lc],
                             (uint32_t)(uintptr_t)&As[1][lr * LDSK + lc] };
    const uint32_t lB[2] = { (uint32_t)(uintptr_t)&Bs[0][lr * LDSK + lc],
                             (uint32_t)(uintptr_t)&Bs[1][lr * LDSK + lc] };

    auto issue = [&](int kt, int buf) {
        const uint64_t a = gA + (uint64_t)kt * (BK * 2);   // bytes
        const uint64_t b = gB + (uint64_t)kt * (BK * 2);
        async_ld16(lA[buf],      a);
        async_ld16(lA[buf] + 16, a + 16);
        async_ld16(lB[buf],      b);
        async_ld16(lB[buf] + 16, b + 16);
    };

    const int nk = K / BK;
    issue(0, 0);                              // prologue: tile 0 in flight
    for (int kt = 0; kt < nk; ++kt) {
        __syncthreads();                      // prev compute on other buffer done
        if (kt + 1 < nk) {
            issue(kt + 1, (kt + 1) & 1);      // keep next tile in flight
            wait_async_le4();                 // tile kt landed (in-order per wave)
        } else {
            wait_async_le0();
        }
        __syncthreads();                      // all waves' tile-kt data visible

        const bf16_t* Ab = As[kt & 1];
        const bf16_t* Bb = Bs[kt & 1];
        bf16x16 af[4], bfr[2];
#pragma unroll
        for (int i = 0; i < 4; ++i) af[i]  = ld_frag(Ab, wm + i * 16, LDSK, lane);
#pragma unroll
        for (int j = 0; j < 2; ++j) bfr[j] = ld_frag(Bb, wn + j * 16, LDSK, lane);
#pragma unroll
        for (int i = 0; i < 4; ++i)
#pragma unroll
            for (int j = 0; j < 2; ++j)
                acc[i][j] = wmma_bf16(af[i], bfr[j], acc[i][j]);
    }

    // epilogue: C/D layout — VGPR r: M = r + 8*(lane>=16), N = lane&15
    const int mb = (lane >> 4) * 8;
    const int nl = lane & 15;
#pragma unroll
    for (int i = 0; i < 4; ++i)
#pragma unroll
        for (int j = 0; j < 2; ++j) {
            const int gn = bn + wn + j * 16 + nl;
#pragma unroll
            for (int r = 0; r < 8; ++r) {
                const int gm = bm + wm + i * 16 + mb + r;
                const size_t idx = (size_t)gm * N + gn;
                const float v = acc[i][j][r];
                if (EPI == EPI_F32)           outF[idx] = v;
                else if (EPI == EPI_F32_BIAS) outF[idx] = v + bias[gn];
                else if (EPI == EPI_BF16)     outB[idx] = (bf16_t)v;
                else if (EPI == EPI_RES_ADD)  outF[idx] = res[idx] + v;
                else                          outF[idx] += v;
            }
        }
}

// ---------------------------------------------------------------- attention
// One block per (b,h). L=64, DK=64.
__global__ __launch_bounds__(256) void attn_kernel(
    const bf16_t* __restrict__ q, const bf16_t* __restrict__ k,
    const bf16_t* __restrict__ v,
    const bf16_t* __restrict__ rk, const bf16_t* __restrict__ rv,   // (L,L,DK)
    const float*  __restrict__ sbias,                               // (B,H,L,L)
    bf16_t* __restrict__ out)                                       // (B*L, D)
{
    constexpr int L = 64, DK = 64, D = 1024, STR = 72, SSTR = 66;
    __shared__ __align__(16) bf16_t Qs[L * STR];
    __shared__ __align__(16) bf16_t Ks[L * STR];
    __shared__ __align__(16) bf16_t Vt[DK * STR];   // transposed: rows=d, cols=j
    __shared__ __align__(16) bf16_t Ps[L * STR];
    __shared__ float Sc[L * SSTR];

    const int bh = blockIdx.x, b = bh >> 4, h = bh & 15;
    const int tid = threadIdx.x, lane = tid & 31, w = tid >> 5;

    // stage Q, K via async global->LDS; V transposed via registers
    {
        const int r = tid >> 2, c = (tid & 3) * 16;
        const size_t go = ((size_t)(b * L + r)) * D + h * DK + c;
        const uint64_t gq = (uint64_t)(uintptr_t)(q + go);
        const uint64_t gk = (uint64_t)(uintptr_t)(k + go);
        async_ld16((uint32_t)(uintptr_t)&Qs[r * STR + c],     gq);
        async_ld16((uint32_t)(uintptr_t)&Qs[r * STR + c + 8], gq + 16);
        async_ld16((uint32_t)(uintptr_t)&Ks[r * STR + c],     gk);
        async_ld16((uint32_t)(uintptr_t)&Ks[r * STR + c + 8], gk + 16);
        bf16_t tmp[16];
        *(uint4*)&tmp[0] = *(const uint4*)(v + go);
        *(uint4*)&tmp[8] = *(const uint4*)(v + go + 8);
#pragma unroll
        for (int e = 0; e < 16; ++e) Vt[(c + e) * STR + r] = tmp[e];
        wait_async_le0();
    }
    __syncthreads();

    // content scores: 16 (4x4) WMMA tiles, 2 per wave
    {
        const int mb = (lane >> 4) * 8, nl = lane & 15;
        for (int t = w * 2; t < w * 2 + 2; ++t) {
            const int ti = (t >> 2) * 16, tj = (t & 3) * 16;
            f32x8 acc = zero8();
#pragma unroll
            for (int kk = 0; kk < DK; kk += 32)
                acc = wmma_bf16(ld_frag(Qs + kk, ti, STR, lane),
                                ld_frag(Ks + kk, tj, STR, lane), acc);
#pragma unroll
            for (int r = 0; r < 8; ++r)
                Sc[(ti + mb + r) * SSTR + tj + nl] = acc[r];
        }
    }
    __syncthreads();

    // relative-K term + scale + smolgen bias (VALU; ~1K MACs/thread)
    {
        const int i = tid >> 2, j0 = (tid & 3) * 16;
        for (int j = j0; j < j0 + 16; ++j) {
            const bf16_t* rp = rk + ((size_t)(i * L + j)) * DK;
            float dot = 0.f;
#pragma unroll 8
            for (int d = 0; d < DK; ++d)
                dot += (float)Qs[i * STR + d] * (float)rp[d];
            const float s = (Sc[i * SSTR + j] + dot) * 0.125f +
                            sbias[(((size_t)bh) * L + i) * L + j];
            Sc[i * SSTR + j] = s;
        }
    }
    __syncthreads();

    // softmax rows
    if (tid < L) {
        float mx = -3.0e38f;
        for (int j = 0; j < L; ++j) mx = fmaxf(mx, Sc[tid * SSTR + j]);
        float sum = 0.f;
        for (int j = 0; j < L; ++j) {
            const float e = __expf(Sc[tid * SSTR + j] - mx);
            Sc[tid * SSTR + j] = e;
            sum += e;
        }
        const float inv = 1.f / sum;
        for (int j = 0; j < L; ++j)
            Ps[tid * STR + j] = (bf16_t)(Sc[tid * SSTR + j] * inv);
    }
    __syncthreads();

    // out = P @ V (WMMA) + P @ r_v (VALU, i-dependent) -> global bf16
    {
        const int mb = (lane >> 4) * 8, nl = lane & 15;
        for (int t = w * 2; t < w * 2 + 2; ++t) {
            const int ti = (t >> 2) * 16, td = (t & 3) * 16;
            f32x8 acc = zero8();
#pragma unroll
            for (int kk = 0; kk < L; kk += 32)
                acc = wmma_bf16(ld_frag(Ps + kk, ti, STR, lane),
                                ld_frag(Vt + kk, td, STR, lane), acc);
#pragma unroll
            for (int r = 0; r < 8; ++r) {
                const int ii = ti + mb + r, dd = td + nl;
                const bf16_t* rp = rv + ((size_t)ii * L) * DK + dd;  // stride DK over j
                float relv = 0.f;
#pragma unroll 8
                for (int j = 0; j < L; ++j)
                    relv += (float)Ps[ii * STR + j] * (float)rp[(size_t)j * DK];
                out[((size_t)(b * L + ii)) * D + h * DK + dd] = (bf16_t)(acc[r] + relv);
            }
        }
    }
}

// ---------------------------------------------------------------- norms etc.
__global__ __launch_bounds__(256) void adaln_rms_kernel(
    const float* __restrict__ x, const float* __restrict__ mod,
    const float* __restrict__ scale, int mod_off, bf16_t* __restrict__ out)
{
    constexpr int D = 1024;
    __shared__ float red[256];
    const int row = blockIdx.x, b = row >> 6;
    const float* xr = x + (size_t)row * D;
    float ss = 0.f;
    for (int d = threadIdx.x; d < D; d += 256) { const float u = xr[d]; ss += u * u; }
    red[threadIdx.x] = ss;
    __syncthreads();
    for (int s = 128; s > 0; s >>= 1) {
        if (threadIdx.x < s) red[threadIdx.x] += red[threadIdx.x + s];
        __syncthreads();
    }
    const float rinv = rsqrtf(red[0] * (1.0f / D) + 1e-6f);
    const float* mrow = mod + (size_t)b * 4096 + mod_off;
    bf16_t* orow = out + (size_t)row * D;
    for (int d = threadIdx.x; d < D; d += 256) {
        float u = xr[d] * rinv * scale[d];
        u = u * (1.f + mrow[d]) + mrow[D + d];
        orow[d] = (bf16_t)u;
    }
}

__global__ void convT_kernel(const float* __restrict__ in, bf16_t* __restrict__ out,
                             int K, int N) {   // in: KxN f32 -> out: NxK bf16
    const size_t total = (size_t)K * N;
    for (size_t i = (size_t)blockIdx.x * blockDim.x + threadIdx.x; i < total;
         i += (size_t)gridDim.x * blockDim.x) {
        const int kk = (int)(i / N), nn = (int)(i % N);
        out[(size_t)nn * K + kk] = (bf16_t)in[i];
    }
}

__global__ void conv_kernel(const float* __restrict__ in, bf16_t* __restrict__ out, size_t n) {
    for (size_t i = (size_t)blockIdx.x * blockDim.x + threadIdx.x; i < n;
         i += (size_t)gridDim.x * blockDim.x)
        out[i] = (bf16_t)in[i];
}

__global__ void relgather_kernel(const int* __restrict__ ridx,
                                 const float* __restrict__ rk, const float* __restrict__ rv,
                                 bf16_t* __restrict__ ork, bf16_t* __restrict__ orv) {
    const int idx = blockIdx.x * 256 + threadIdx.x;   // 64*64*64 = 262144
    if (idx >= 64 * 64 * 64) return;
    const int d = idx & 63, ij = idx >> 6;
    const int r = ridx[ij];
    ork[idx] = (bf16_t)rk[r * 64 + d];
    orv[idx] = (bf16_t)rv[r * 64 + d];
}

__global__ void swiglu_kernel(const bf16_t* __restrict__ g1, const bf16_t* __restrict__ g2,
                              bf16_t* __restrict__ ff, size_t n) {
    for (size_t i = (size_t)blockIdx.x * blockDim.x + threadIdx.x; i < n;
         i += (size_t)gridDim.x * blockDim.x) {
        const float a = (float)g1[i];
        const float s = a / (1.f + __expf(-a));
        ff[i] = (bf16_t)(s * (float)g2[i]);
    }
}

// ---------------------------------------------------------------- launch
extern "C" void kernel_launch(void* const* d_in, const int* in_sizes, int n_in,
                              void* d_out, int out_size, void* d_ws, size_t ws_size,
                              hipStream_t stream) {
    constexpr int B = 512, L = 64, D = 1024, H = 16, DK = 64, DFF = 2816, CTX = 256;
    constexpr int M = B * L;                   // 32768

    const float* x      = (const float*)d_in[0];
    const int*   ridx   = (const int*)  d_in[1];
    const float* sbias  = (const float*)d_in[2];
    const float* ctx    = (const float*)d_in[3];
    const float* n1s    = (const float*)d_in[4];
    const float* n2s    = (const float*)d_in[5];
    const float* wq     = (const float*)d_in[6];
    const float* wk     = (const float*)d_in[7];
    const float* wv     = (const float*)d_in[8];
    const float* wo     = (const float*)d_in[9];
    const float* relk   = (const float*)d_in[10];
    const float* relv   = (const float*)d_in[11];
    const float* w1     = (const float*)d_in[12];
    const float* w2     = (const float*)d_in[13];
    const float* w3     = (const float*)d_in[14];
    const float* adaw   = (const float*)d_in[15];
    const float* adab   = (const float*)d_in[16];
    float* out = (float*)d_out;

    // workspace bump allocator (256B aligned)
    char* ws = (char*)d_ws;
    size_t off = 0;
    auto alloc = [&](size_t bytes) -> char* {
        char* p = ws + off;
        off += (bytes + 255) & ~(size_t)255;
        return p;
    };
    const size_t SZ_DD  = (size_t)D * D * 2;        // 2 MB
    const size_t SZ_DF  = (size_t)D * DFF * 2;      // 5.77 MB
    bf16_t* wqT  = (bf16_t*)alloc(SZ_DD);
    bf16_t* wkT  = (bf16_t*)alloc(SZ_DD);
    bf16_t* wvT  = (bf16_t*)alloc(SZ_DD);
    bf16_t* woT  = (bf16_t*)alloc(SZ_DD);
    bf16_t* w1T  = (bf16_t*)alloc(SZ_DF);
    bf16_t* w2T  = (bf16_t*)alloc(SZ_DF);
    bf16_t* w3T  = (bf16_t*)alloc(SZ_DF);
    bf16_t* adaT = (bf16_t*)alloc((size_t)4 * D * CTX * 2);
    bf16_t* ctxB = (bf16_t*)alloc((size_t)B * CTX * 2);
    bf16_t* rkB  = (bf16_t*)alloc((size_t)L * L * DK * 2);
    bf16_t* rvB  = (bf16_t*)alloc((size_t)L * L * DK * 2);
    float*  mod  = (float*) alloc((size_t)B * 4 * D * 4);
    bf16_t* hB   = (bf16_t*)alloc((size_t)M * D * 2);
    // phase-overlapped arena: attention (q,k,v,attnout) vs FFN (g1,g2,ff)
    const size_t SZ_MD  = (size_t)M * D * 2;        // 67 MB
    const size_t SZ_MF  = (size_t)M * DFF * 2;      // 184.5 MB
    char* arena = alloc(3 * SZ_MF);
    bf16_t* qB  = (bf16_t*)(arena);
    bf16_t* kB  = (bf16_t*)(arena + SZ_MD);
    bf16_t* vB  = (bf16_t*)(arena + 2 * SZ_MD);
    bf16_t* aoB = (bf16_t*)(arena + 3 * SZ_MD);
    bf16_t* g1B = (bf16_t*)(arena);
    bf16_t* g2B = (bf16_t*)(arena + SZ_MF);
    bf16_t* ffB = (bf16_t*)(arena + 2 * SZ_MF);

    // ---- weight prep (bf16 transposes) ----
    convT_kernel<<<4096, 256, 0, stream>>>(wq, wqT, D, D);
    convT_kernel<<<4096, 256, 0, stream>>>(wk, wkT, D, D);
    convT_kernel<<<4096, 256, 0, stream>>>(wv, wvT, D, D);
    convT_kernel<<<4096, 256, 0, stream>>>(wo, woT, D, D);
    convT_kernel<<<8192, 256, 0, stream>>>(w1, w1T, D, DFF);
    convT_kernel<<<8192, 256, 0, stream>>>(w2, w2T, D, DFF);
    convT_kernel<<<8192, 256, 0, stream>>>(w3, w3T, DFF, D);
    convT_kernel<<<4096, 256, 0, stream>>>(adaw, adaT, CTX, 4 * D);
    conv_kernel<<<512, 256, 0, stream>>>(ctx, ctxB, (size_t)B * CTX);
    relgather_kernel<<<1024, 256, 0, stream>>>(ridx, relk, relv, rkB, rvB);

    // ---- AdaLN modulation GEMM: mod = ctx @ ada_w + ada_b ----
    gemm_bf16_kernel<EPI_F32_BIAS><<<dim3(4 * D / BN, B / BM), 256, 0, stream>>>(
        ctxB, adaT, B, 4 * D, CTX, mod, nullptr, adab, nullptr);

    // ---- attention sub-block ----
    adaln_rms_kernel<<<M, 256, 0, stream>>>(x, mod, n1s, 0, hB);
    gemm_bf16_kernel<EPI_BF16><<<dim3(D / BN, M / BM), 256, 0, stream>>>(
        hB, wqT, M, D, D, nullptr, qB, nullptr, nullptr);
    gemm_bf16_kernel<EPI_BF16><<<dim3(D / BN, M / BM), 256, 0, stream>>>(
        hB, wkT, M, D, D, nullptr, kB, nullptr, nullptr);
    gemm_bf16_kernel<EPI_BF16><<<dim3(D / BN, M / BM), 256, 0, stream>>>(
        hB, wvT, M, D, D, nullptr, vB, nullptr, nullptr);
    attn_kernel<<<B * H, 256, 0, stream>>>(qB, kB, vB, rkB, rvB, sbias, aoB);
    gemm_bf16_kernel<EPI_RES_ADD><<<dim3(D / BN, M / BM), 256, 0, stream>>>(
        aoB, woT, M, D, D, out, nullptr, nullptr, x);   // out = x + attn

    // ---- SwiGLU FFN sub-block ----
    adaln_rms_kernel<<<M, 256, 0, stream>>>(out, mod, n2s, 2 * D, hB);
    gemm_bf16_kernel<EPI_BF16><<<dim3(DFF / BN, M / BM), 256, 0, stream>>>(
        hB, w1T, M, DFF, D, nullptr, g1B, nullptr, nullptr);
    gemm_bf16_kernel<EPI_BF16><<<dim3(DFF / BN, M / BM), 256, 0, stream>>>(
        hB, w2T, M, DFF, D, nullptr, g2B, nullptr, nullptr);
    swiglu_kernel<<<4096, 256, 0, stream>>>(g1B, g2B, ffB, (size_t)M * DFF);
    gemm_bf16_kernel<EPI_ACCUM><<<dim3(D / BN, M / BM), 256, 0, stream>>>(
        ffB, w3T, M, D, DFF, out, nullptr, nullptr, nullptr);  // out += ffn
    (void)n_in; (void)in_sizes; (void)out_size; (void)ws_size;
}
